// BFConv2d_91336774517269
// MI455X (gfx1250) — matrix operand order repaired and checked by reading, not compiled, not run on hardware
//
#include <hip/hip_runtime.h>

// ---------------------------------------------------------------------------
// BFP-quantized 3x3 conv as implicit GEMM on CDNA5 (gfx1250, wave32).
//   GEMM: M = N*H*W = 50176, N = Cout = 256, K = 2304 with ordering
//   k = (kh*3+kw)*256 + cin  -> each KT=64 tile sits inside one filter tap,
//   making the im2col gather a strided copy with uniform predicates.
//   Weights are BFP-quantized AND repacked to B-layout [tap][cin][cout] so
//   every B tile is a dense 2D tile -> staged by the Tensor Data Mover.
// BFP values are int8-mantissa * 2^e -> exact in FP16 -> f16 WMMA path.
// ---------------------------------------------------------------------------

typedef _Float16 half_t;
typedef _Float16 v16h  __attribute__((ext_vector_type(16)));
typedef _Float16 v8h   __attribute__((ext_vector_type(8)));
typedef float    v8f   __attribute__((ext_vector_type(8)));
typedef unsigned int u32x4 __attribute__((ext_vector_type(4)));
typedef int      i32x4 __attribute__((ext_vector_type(4)));
typedef int      i32x8 __attribute__((ext_vector_type(8)));

#define GROUP   36
#define IN_C    256
#define IMG_HW  56
#define PIX     (IMG_HW * IMG_HW)          // 3136
#define C_OUT   256
#define GEMM_K  (IN_C * 9)                 // 2304
#define BM      128
#define BN      64
#define KT      64
#define LDA     (KT + 8)                   // halves; 144 B rows (16B aligned)

#if __has_builtin(__builtin_amdgcn_tensor_load_to_lds)
#define USE_TDM 1
#else
#define USE_TDM 0
#endif
#if __has_include(<hip/amd_detail/amd_gfx1250_TDM.h>)
#define TDM_SIX_ARGS 1
#else
#define TDM_SIX_ARGS 0
#endif

// -------------------------- BFP quantize x -> fp16 (NCHW) ------------------
__global__ void __launch_bounds__(256)
bfp_quant_x(const float* __restrict__ src, half_t* __restrict__ dst, int n)
{
    int g = blockIdx.x * blockDim.x + threadIdx.x;
    long base = (long)g * GROUP;
    if (base >= n) return;
    int cnt = n - (int)base;
    if (cnt > GROUP) cnt = GROUP;

    float v[GROUP];
    float maxabs = 0.0f;
    for (int j = 0; j < cnt; ++j) {
        v[j] = src[base + j];
        maxabs = fmaxf(maxabs, fabsf(v[j]));
    }
    float e     = floorf(log2f(fmaxf(maxabs, 1e-38f)));
    float scale = exp2f(e - 7.0f);
    float inv   = 1.0f / scale;              // exact: power of two
    for (int j = 0; j < cnt; ++j) {
        float q = rintf(v[j] * inv);         // round-half-even == jnp.round
        q = fminf(fmaxf(q, -128.0f), 127.0f) * scale;
        if (maxabs == 0.0f) q = 0.0f;
        dst[base + j] = (half_t)q;
    }
}

// ---------- BFP quantize w (groups in flat OIHW) -> repack [tap][cin][cout]
__global__ void __launch_bounds__(256)
bfp_quant_w_repack(const float* __restrict__ src, half_t* __restrict__ dst, int n)
{
    int g = blockIdx.x * blockDim.x + threadIdx.x;
    long base = (long)g * GROUP;
    if (base >= n) return;
    int cnt = n - (int)base;
    if (cnt > GROUP) cnt = GROUP;

    float v[GROUP];
    float maxabs = 0.0f;
    for (int j = 0; j < cnt; ++j) {
        v[j] = src[base + j];
        maxabs = fmaxf(maxabs, fabsf(v[j]));
    }
    float e     = floorf(log2f(fmaxf(maxabs, 1e-38f)));
    float scale = exp2f(e - 7.0f);
    float inv   = 1.0f / scale;
    for (int j = 0; j < cnt; ++j) {
        float q = rintf(v[j] * inv);
        q = fminf(fmaxf(q, -128.0f), 127.0f) * scale;
        if (maxabs == 0.0f) q = 0.0f;
        int f   = (int)base + j;             // flat OIHW index
        int o   = f / GEMM_K;
        int r   = f - o * GEMM_K;
        int ci  = r / 9;
        int tap = r - ci * 9;                // kh*3 + kw
        dst[((size_t)tap * IN_C + ci) * C_OUT + o] = (half_t)q;
    }
}

// ------------------------- implicit-GEMM conv (WMMA + TDM) -----------------
__global__ void __launch_bounds__(256)
conv_wmma(const half_t* __restrict__ xq, const half_t* __restrict__ wq2,
          const float* __restrict__ bias, float* __restrict__ out)
{
    __shared__ half_t sA[BM * LDA];          // A tile [BM][KT] (padded rows)
    __shared__ half_t sB[KT * BN];           // B tile [KT][BN], packed (TDM)

    const int tid   = threadIdx.x;
    const int lane  = tid & 31;
    const int wave  = tid >> 5;
    const int waveM = wave & 3;              // 4 waves along M
    const int waveN = wave >> 2;             // 2 waves along N
    const int bm    = blockIdx.y * BM;
    const int bn    = blockIdx.x * BN;

    // k0-invariant im2col coordinates: thread owns one 32-wide run of one row
    const int arow  = tid >> 1;              // 0..127
    const int khalf = (tid & 1) << 5;        // 0 or 32
    const int gm    = bm + arow;
    const int nb    = gm / PIX;
    const int rem   = gm - nb * PIX;
    const int h     = rem / IMG_HW;
    const int w     = rem - h * IMG_HW;
    const half_t* xn = xq + (size_t)nb * IN_C * PIX;
    half_t* sArow = &sA[arow * LDA + khalf];

#if USE_TDM
    const unsigned sB_lds = (unsigned)(size_t)(&sB[0]);  // LDS byte offset
#endif

    v8f acc[2][2] = {};

    for (int k0 = 0; k0 < GEMM_K; k0 += KT) {
        const int tap = k0 >> 8;             // uniform per tile (2304 = 9*256)
        const int c0  = k0 & 255;
        const int kh  = tap / 3;
        const int kw  = tap - kh * 3;
        const int ih  = h + kh - 1;
        const int iw  = w + kw - 1;
        const bool valid = ((unsigned)ih < (unsigned)IMG_HW) &&
                           ((unsigned)iw < (unsigned)IMG_HW);
        const half_t* abase = xn + (size_t)(c0 + khalf) * PIX + ih * IMG_HW + iw;

        // ---- stage A: 32 channel-strided fp16 loads, uniform predicate -----
        #pragma unroll
        for (int j = 0; j < 32; ++j) {
            half_t v = (half_t)0.0f;
            if (valid) v = abase[j * PIX];
            sArow[j] = v;
        }
        if (valid && c0 < 192)               // prefetch next K-tile of this tap
            __builtin_prefetch(abase + 64 * PIX, 0, 0);

        // ---- stage B: 64x64 2D tile of wq2[2304][256] via Tensor Data Mover
#if USE_TDM
        if (wave == 0) {
            unsigned long long ga =
                (unsigned long long)(size_t)(wq2 + (size_t)k0 * C_OUT + bn);
            u32x4 g0;
            g0[0] = 1u;                                   // count=1, user mode
            g0[1] = sB_lds;                               // lds_addr
            g0[2] = (unsigned)(ga & 0xFFFFFFFFu);         // global_addr[31:0]
            g0[3] = (unsigned)((ga >> 32) & 0x01FFFFFFu)  // global_addr[56:32]
                    | (2u << 30);                         // type=2 (image)
            i32x8 g1;
            g1[0] = 1 << 16;                              // data_size=2B
            g1[1] = (int)((unsigned)C_OUT << 16);         // tensor_dim0 lo16
            g1[2] = (int)((unsigned)GEMM_K << 16);        // dim0 hi=0|dim1 lo16
            g1[3] = (int)((unsigned)KT << 16);            // dim1 hi=0|tile_dim0
            g1[4] = KT;                                   // tile_dim1 (tile_dim2=0)
            g1[5] = C_OUT;                                // tensor_dim0_stride lo32
            g1[6] = 0;                                    // stride hi | dim1_stride lo
            g1[7] = 0;
            i32x4 gz = {0, 0, 0, 0};                      // groups 2/3 unused (2D)
#if TDM_SIX_ARGS
            i32x8 gz8 = {0, 0, 0, 0, 0, 0, 0, 0};
            __builtin_amdgcn_tensor_load_to_lds(g0, g1, gz, gz, gz8, 0);
#else
            __builtin_amdgcn_tensor_load_to_lds(g0, g1, gz, gz, 0);
#endif
        }
#else
        for (int i = tid; i < KT * BN; i += 256) {
            int kk = i >> 6, col = i & 63;
            sB[kk * BN + col] = wq2[(size_t)(k0 + kk) * C_OUT + bn + col];
        }
#endif

#if USE_TDM
        if (wave == 0) __builtin_amdgcn_s_wait_tensorcnt(0);
#endif
        __syncthreads();

        // ---- compute: 2 WMMA K-steps of 32 per LDS tile --------------------
        #pragma unroll
        for (int ks = 0; ks < KT; ks += 32) {
            // A 16x32 f16 layout: lanes 0-15 K 0-7/16-23, lanes 16-31 K 8-15/24-31
            const int kbA = ks + ((lane < 16) ? 0 : 8);
            v16h a[2], b[2];
            #pragma unroll
            for (int t = 0; t < 2; ++t) {
                int row = waveM * 32 + t * 16 + (lane & 15);
                const half_t* p = &sA[row * LDA + kbA];
                v8h lo = *(const v8h*)(p);        // K kbA .. kbA+7
                v8h hi = *(const v8h*)(p + 16);   // K kbA+16 .. kbA+23
                a[t] = __builtin_shufflevector(lo, hi,
                        0, 1, 2, 3, 4, 5, 6, 7, 8, 9, 10, 11, 12, 13, 14, 15);
            }
            // B 32x16: lanes 0-15 hold K 0-7/16-23, lanes 16-31 K 8-15/24-31, N = lane&15
            const int kbB = ks + ((lane >> 4) << 3);
            #pragma unroll
            for (int t = 0; t < 2; ++t) {
                int col = waveN * 32 + t * 16 + (lane & 15);
                #pragma unroll
                for (int h2 = 0; h2 < 16; ++h2)
                    b[t][h2] = sB[(kbB + h2 + ((h2 >> 3) << 3)) * BN + col];
            }
            #pragma unroll
            for (int ti = 0; ti < 2; ++ti)
                #pragma unroll
                for (int tj = 0; tj < 2; ++tj)
                    acc[ti][tj] = __builtin_amdgcn_wmma_f32_16x16x32_f16(
                        false, a[ti], false, b[tj],
                        (short)0, acc[ti][tj], false, false);
        }
        __syncthreads();
    }

    // ---- epilogue: C layout VGPR v -> M = v (+8 lanes 16-31), N = lane&15 --
    #pragma unroll
    for (int ti = 0; ti < 2; ++ti) {
        #pragma unroll
        for (int tj = 0; tj < 2; ++tj) {
            int col = bn + waveN * 32 + tj * 16 + (lane & 15);
            float bv = bias[col];
            #pragma unroll
            for (int v = 0; v < 8; ++v) {
                int row = bm + waveM * 32 + ti * 16 + ((lane < 16) ? v : v + 8);
                int n2  = row / PIX;
                int r2  = row - n2 * PIX;
                out[((size_t)n2 * C_OUT + col) * PIX + r2] = acc[ti][tj][v] + bv;
            }
        }
    }
}

// ---------------------------------------------------------------------------
extern "C" void kernel_launch(void* const* d_in, const int* in_sizes, int n_in,
                              void* d_out, int out_size, void* d_ws, size_t ws_size,
                              hipStream_t stream)
{
    const float* x    = (const float*)d_in[0];   // 16*256*56*56
    const float* w    = (const float*)d_in[1];   // 256*256*3*3
    const float* bias = (const float*)d_in[2];   // 256
    float*       out  = (float*)d_out;

    const int nx = in_sizes[0];
    const int nw = in_sizes[1];

    half_t* xq  = (half_t*)d_ws;
    half_t* wq2 = xq + (size_t)nx;               // [tap][cin][cout]

    int ngx = (nx + GROUP - 1) / GROUP;
    int ngw = (nw + GROUP - 1) / GROUP;
    bfp_quant_x       <<<(ngx + 255) / 256, 256, 0, stream>>>(x, xq, nx);
    bfp_quant_w_repack<<<(ngw + 255) / 256, 256, 0, stream>>>(w, wq2, nw);

    dim3 grid(C_OUT / BN, (16 * PIX) / BM);      // (4, 392)
    conv_wmma<<<grid, 256, 0, stream>>>(xq, wq2, bias, out);
}
